// Attention_78434692760060
// MI455X (gfx1250) — matrix operand order repaired and checked by reading, not compiled, not run on hardware
//
#include <hip/hip_runtime.h>
#include <hip/hip_bf16.h>
#include <math.h>

#define B_ 32
#define T_ 2048
#define N_ 1024

typedef __attribute__((ext_vector_type(16))) __bf16 v16bf;
typedef __attribute__((ext_vector_type(8)))  __bf16 v8bf;
typedef __attribute__((ext_vector_type(4)))  __bf16 v4bf;
typedef __attribute__((ext_vector_type(8)))  float  v8f;

// 1024 bf16 + 8 pad -> row stride 2064 B = 516 dwords -> successive rows land
// 4 banks apart, so the 16 lanes' ds_load_b128 A-fragment reads cover all 64 banks.
#define LDSA_STRIDE 1032

// ---------------------------------------------------------------------------
// Kernel 0a: dec_fea[b,m] = s_t_hat[b,:] . W_dec[m,:] + b_dec[m]   (67 MFLOP)
// ---------------------------------------------------------------------------
__global__ __launch_bounds__(256) void dec_features_kernel(
    const float* __restrict__ s_t_hat, const float* __restrict__ Wdec,
    const float* __restrict__ bdec, float* __restrict__ dec_fea)
{
    const int idx = blockIdx.x * 256 + threadIdx.x;   // 0 .. B*N-1
    const int b = idx >> 10;
    const int m = idx & (N_ - 1);
    const float* srow = s_t_hat + (size_t)b * N_;
    const float* wrow = Wdec    + (size_t)m * N_;
    float ax = 0.f, ay = 0.f, az = 0.f, aw = 0.f;
    for (int n = 0; n < N_; n += 4) {
        float4 sv = *(const float4*)(srow + n);
        float4 wv = *(const float4*)(wrow + n);
        ax += sv.x * wv.x; ay += sv.y * wv.y;
        az += sv.z * wv.z; aw += sv.w * wv.w;
    }
    dec_fea[idx] = (ax + ay) + (az + aw) + bdec[m];
}

// ---------------------------------------------------------------------------
// Kernel 0b: prepack W_h (f32) -> bf16, same row-major layout. 4 MB -> 2 MB,
// L2-resident for the whole GEMM; removes f32->bf16 cvt from the hot loop.
// ---------------------------------------------------------------------------
__global__ __launch_bounds__(256) void pack_wh_kernel(
    const float* __restrict__ Wh, __bf16* __restrict__ Whb)
{
    const int i = (blockIdx.x * 256 + threadIdx.x) * 4;
    float4 f = *(const float4*)(Wh + i);
    v4bf p;
    p[0] = (__bf16)f.x; p[1] = (__bf16)f.y; p[2] = (__bf16)f.z; p[3] = (__bf16)f.w;
    *(v4bf*)(Whb + i) = p;
}

// ---------------------------------------------------------------------------
// Kernel 1: fused scores[b,t] = v . tanh(h[b,t,:]@Wh^T + dec_fea[b,:] +
//                                        coverage[b,t]*Wc)
// One workgroup = (b, 16-row t-tile), 4 waves; wave w sweeps m-tiles
// w*16, w*16+64, ...  K-loop: V_WMMA_F32_16X16X32_BF16.
// ---------------------------------------------------------------------------
__global__ __launch_bounds__(128) void attn_scores_wmma_kernel(
    const float*  __restrict__ h, const __bf16* __restrict__ Whb,
    const float*  __restrict__ dec_fea, const float* __restrict__ coverage,
    const float*  __restrict__ Wc, const float* __restrict__ vvec,
    float* __restrict__ scores)
{
    __shared__ __align__(16) __bf16 ldsA[16 * LDSA_STRIDE];
    __shared__ float scoreLds[16];

    const int b    = blockIdx.y;
    const int t0   = blockIdx.x * 16;
    const int tid  = threadIdx.x;          // 0..127 = 4 waves (wave32)
    const int wave = tid >> 5;
    const int lane = tid & 31;

    if (tid < 16) scoreLds[tid] = 0.0f;

    // Stage 16 rows of h as bf16 into LDS (read from HBM exactly once).
    const float* hblk = h + ((size_t)b * T_ + t0) * N_;
    for (int idx = tid; idx < 16 * (N_ / 4); idx += 128) {
        const int r  = idx >> 8;              // N_/4 == 256 float4 per row
        const int c4 = (idx & 255) << 2;
        float4 f = *(const float4*)(hblk + (size_t)r * N_ + c4);
        v4bf p;
        p[0] = (__bf16)f.x; p[1] = (__bf16)f.y; p[2] = (__bf16)f.z; p[3] = (__bf16)f.w;
        *(v4bf*)(&ldsA[r * LDSA_STRIDE + c4]) = p;
    }
    __syncthreads();

    // wave32 WMMA fragment geometry (this kernel's contiguous k runs, same for A and B):
    const int laneHi = lane >> 4;      // 0: lanes 0-15, 1: lanes 16-31
    const int lmod   = lane & 15;
    const int kOffA  = laneHi * 16;    // A run : k0+kOffA..+15
    const int kOffB  = laneHi * 16;    // B run : k0+kOffB..+15
    const int rowOff = laneHi * 8;     // C/D  : M = r + rowOff

    float cov[8];
    #pragma unroll
    for (int r = 0; r < 8; ++r)
        cov[r] = coverage[(size_t)b * T_ + t0 + r + rowOff];

    float scoreAcc[8];
    #pragma unroll
    for (int r = 0; r < 8; ++r) scoreAcc[r] = 0.0f;

    for (int m0 = wave * 16; m0 < N_; m0 += 4 * 16) {
        const int m = m0 + lmod;
        const __bf16* wrow = Whb + (size_t)m * N_;
        v8f acc = {};
        for (int k0 = 0; k0 < N_; k0 += 32) {
            // A fragment from LDS (2x ds_load_b128)
            const __bf16* ap = &ldsA[lmod * LDSA_STRIDE + k0 + kOffA];
            v8bf alo = *(const v8bf*)(ap);
            v8bf ahi = *(const v8bf*)(ap + 8);
            v16bf A = __builtin_shufflevector(alo, ahi,
                        0, 1, 2, 3, 4, 5, 6, 7, 8, 9, 10, 11, 12, 13, 14, 15);
            // B fragment (Wh^T tile) straight from L2-resident packed Wh
            v16bf Bm = *(const v16bf*)(wrow + k0 + kOffB);
            // D = A x B + C, f32 accumulate
            acc = __builtin_amdgcn_wmma_f32_16x16x32_bf16(
                      false, A, false, Bm, (short)0, acc, false, false);
        }
        // Fused epilogue: + dec_fea + cov*Wc, tanh, dot with v (per-column m)
        const float dec = dec_fea[(size_t)b * N_ + m];
        const float wc  = Wc[m];
        const float vm  = vvec[m];
        #pragma unroll
        for (int r = 0; r < 8; ++r) {
            const float e = tanhf(acc[r] + dec + cov[r] * wc);
            scoreAcc[r] += e * vm;
        }
    }
    // Reduce the 16 m-columns per half-wave (and across waves) via LDS atomics.
    #pragma unroll
    for (int r = 0; r < 8; ++r)
        atomicAdd(&scoreLds[r + rowOff], scoreAcc[r]);
    __syncthreads();
    if (tid < 16)
        scores[(size_t)b * T_ + t0 + tid] = scoreLds[tid];
}

// ---------------------------------------------------------------------------
// Kernel 2: masked softmax over T + renorm + coverage update. One WG per b.
// attn = exp(s-max)*mask / sum(exp(s-max)*mask)  (softmax denom cancels)
// ---------------------------------------------------------------------------
__global__ __launch_bounds__(256) void softmax_cov_kernel(
    const float* __restrict__ scores, const float* __restrict__ mask,
    const float* __restrict__ coverage,
    float* __restrict__ attn, float* __restrict__ covnext)
{
    __shared__ float red[256];
    const int b = blockIdx.x, tid = threadIdx.x;
    const size_t base = (size_t)b * T_;

    float vals[8];
    float mx = -3.4e38f;
    #pragma unroll
    for (int i = 0; i < 8; ++i) {
        vals[i] = scores[base + tid + i * 256];
        mx = fmaxf(mx, vals[i]);
    }
    red[tid] = mx; __syncthreads();
    for (int s = 128; s > 0; s >>= 1) {
        if (tid < s) red[tid] = fmaxf(red[tid], red[tid + s]);
        __syncthreads();
    }
    mx = red[0]; __syncthreads();

    float sum = 0.f;
    #pragma unroll
    for (int i = 0; i < 8; ++i) {
        const float e = __expf(vals[i] - mx) * mask[base + tid + i * 256];
        vals[i] = e; sum += e;
    }
    red[tid] = sum; __syncthreads();
    for (int s = 128; s > 0; s >>= 1) {
        if (tid < s) red[tid] += red[tid + s];
        __syncthreads();
    }
    const float inv = 1.0f / red[0];
    #pragma unroll
    for (int i = 0; i < 8; ++i) {
        const size_t ix = base + tid + i * 256;
        const float a = vals[i] * inv;
        attn[ix]    = a;
        covnext[ix] = coverage[ix] + a;
    }
}

// ---------------------------------------------------------------------------
// Kernel 3: c_t[b,n] = sum_t attn[b,t] * h[b,t,n]   (memory-bound, ~11us floor)
// ---------------------------------------------------------------------------
__global__ __launch_bounds__(256) void context_kernel(
    const float* __restrict__ h, const float* __restrict__ attn,
    float* __restrict__ c_t)
{
    __shared__ float aLds[T_];
    const int b = blockIdx.x, tid = threadIdx.x;
    for (int i = tid; i < T_; i += 256) aLds[i] = attn[(size_t)b * T_ + i];
    __syncthreads();
    const int n0 = tid * 4;
    const float* hb = h + (size_t)b * T_ * N_;
    float ax = 0.f, ay = 0.f, az = 0.f, aw = 0.f;
    for (int t = 0; t < T_; ++t) {
        const float a = aLds[t];
        float4 hv = *(const float4*)(hb + (size_t)t * N_ + n0);
        ax += a * hv.x; ay += a * hv.y; az += a * hv.z; aw += a * hv.w;
    }
    float4 o = { ax, ay, az, aw };
    *(float4*)(c_t + (size_t)b * N_ + n0) = o;
}

// ---------------------------------------------------------------------------
extern "C" void kernel_launch(void* const* d_in, const int* in_sizes, int n_in,
                              void* d_out, int out_size, void* d_ws, size_t ws_size,
                              hipStream_t stream)
{
    (void)in_sizes; (void)n_in; (void)out_size; (void)ws_size;

    const float* s_t_hat  = (const float*)d_in[0];
    const float* h        = (const float*)d_in[1];
    const float* mask     = (const float*)d_in[2];
    const float* coverage = (const float*)d_in[3];
    const float* Wh       = (const float*)d_in[4];
    const float* Wdec     = (const float*)d_in[5];
    const float* bdec     = (const float*)d_in[6];
    const float* Wc       = (const float*)d_in[7];
    const float* vvec     = (const float*)d_in[8];

    // d_out: c_t [B*N] | attn_dist [B*T] | coverage_next [B*T]
    float* out     = (float*)d_out;
    float* c_t     = out;
    float* attn    = out + (size_t)B_ * N_;
    float* covnext = attn + (size_t)B_ * T_;

    // d_ws: dec_fea [B*N f32 =128KB] | scores [B*T f32 =256KB] | Whb [N*N bf16 =2MB]
    float*  dec_fea = (float*)d_ws;
    float*  scores  = dec_fea + (size_t)B_ * N_;
    __bf16* Whb     = (__bf16*)(scores + (size_t)B_ * T_);

    dec_features_kernel<<<dim3((B_ * N_) / 256), dim3(256), 0, stream>>>(
        s_t_hat, Wdec, bdec, dec_fea);
    pack_wh_kernel<<<dim3((N_ * N_ / 4) / 256), dim3(256), 0, stream>>>(Wh, Whb);
    attn_scores_wmma_kernel<<<dim3(T_ / 16, B_), dim3(128), 0, stream>>>(
        h, Whb, dec_fea, coverage, Wc, vvec, scores);
    softmax_cov_kernel<<<dim3(B_), dim3(256), 0, stream>>>(
        scores, mask, coverage, attn, covnext);
    context_kernel<<<dim3(B_), dim3(256), 0, stream>>>(h, attn, c_t);
}
